// HyperbolicMessagePassing_84782654423058
// MI455X (gfx1250) — compile-verified, matrix operand order checked by reading, not run on hardware
//
#include <hip/hip_runtime.h>

// ---------------- vector types ----------------
typedef __attribute__((ext_vector_type(4)))  float        f4;
typedef __attribute__((ext_vector_type(4)))  unsigned int u4;
typedef __attribute__((ext_vector_type(8)))  float        v8f;
typedef __attribute__((ext_vector_type(16))) _Float16     v16h;
typedef __attribute__((ext_vector_type(8)))  _Float16     v8h;
typedef __attribute__((ext_vector_type(4)))  _Float16     h4;
typedef int i4v __attribute__((vector_size(16)));   // matches builtin param type

#define DIM    128
#define PITCH  136            // f16 pitch: 272B rows -> 16B aligned, bank-shifted
#define TILE   128            // rows (edges or nodes) per block tile
#define NNODES 50000
#define NEDGES 600000

#define AS1 __attribute__((address_space(1)))
#define AS3 __attribute__((address_space(3)))

// -------- async global->LDS copy (CDNA5 GLOBAL_LOAD_ASYNC_TO_LDS_B128) -----
#if defined(__has_builtin)
#  if __has_builtin(__builtin_amdgcn_global_load_async_to_lds_b128)
#    define HAVE_ASYNC_LDS 1
#  else
#    define HAVE_ASYNC_LDS 0
#  endif
#else
#  define HAVE_ASYNC_LDS 0
#endif

__device__ __forceinline__ void async_cp16(void* lds_dst, const void* gsrc) {
#if HAVE_ASYNC_LDS
    __builtin_amdgcn_global_load_async_to_lds_b128(
        (AS1 i4v*)gsrc, (AS3 i4v*)lds_dst, 0, 0);
#else
    *(u4*)lds_dst = *(const u4*)gsrc;
#endif
}

__device__ __forceinline__ void async_join() {
#if HAVE_ASYNC_LDS
#  if __has_builtin(__builtin_amdgcn_s_wait_asynccnt)
    __builtin_amdgcn_s_wait_asynccnt(0);
#  else
    asm volatile("s_wait_asynccnt 0" ::: "memory");
#  endif
#endif
}

// ---- WMMA fragment load from an LDS row (A and B^T share this packing) ----
// lane L: m/n = L&15, hh = L>>4 ; VGPR j<4 -> k = kt*32 + 8*hh + 2j{+0,1}
//                                  VGPR j>=4 -> k = kt*32 + 16 + 8*hh + ...
__device__ __forceinline__ v16h load_frag(const _Float16* rowbase, int kt, int hh) {
    const _Float16* p = rowbase + kt * 32 + hh * 8;
    union { v16h v; v8h h[2]; } u;
    u.h[0] = *(const v8h*)p;
    u.h[1] = *(const v8h*)(p + 16);
    return u.v;
}

// ---------------- logmap0: x -> artanh(||x||)/||x|| * x  (f16 out) ----------
__global__ void logmap0_kernel(const float* __restrict__ x,
                               _Float16* __restrict__ xt, int n) {
    int wave = threadIdx.x >> 5, lane = threadIdx.x & 31;
    int r = blockIdx.x * 8 + wave;
    if (r >= n) return;
    f4 v = *(const f4*)(x + (size_t)r * DIM + lane * 4);
    float ss = v.x * v.x + v.y * v.y + v.z * v.z + v.w * v.w;
    #pragma unroll
    for (int off = 16; off > 0; off >>= 1) ss += __shfl_xor(ss, off, 32);
    float nrm = sqrtf(ss);
    float nc  = fminf(fmaxf(nrm, 1e-8f), 1.0f - 1e-5f);
    float sc  = (0.5f * logf((1.0f + nc) / (1.0f - nc))) / nc;   // artanh(nc)/nc
    h4 hv = __builtin_convertvector(v * sc, h4);
    *(h4*)(xt + (size_t)r * DIM + lane * 4) = hv;
}

// ---------------- weight transpose fp32 [k][n] -> f16 Wt[n][k] --------------
__global__ void wtrans_kernel(const float* __restrict__ w, _Float16* __restrict__ wt) {
    int idx = blockIdx.x * 256 + threadIdx.x;      // idx = n*128 + k
    int n = idx >> 7, k = idx & 127;
    wt[idx] = (_Float16)w[k * DIM + n];
}

// ---------------- in-degree count -------------------------------------------
__global__ void degree_kernel(const int* __restrict__ row, float* __restrict__ cnt, int e) {
    int i = blockIdx.x * 256 + threadIdx.x;
    if (i < e) atomicAdd(&cnt[row[i]], 1.0f);
}

// ---------------- fused edge MLP: gather -> 2xGEMM -> scatter-add -----------
__global__ void edge_mlp_kernel(const _Float16* __restrict__ xt,
                                const int* __restrict__ ei,   // [0,E)=dst  [E,2E)=src
                                const _Float16* __restrict__ wt1, const float* __restrict__ b1,
                                const _Float16* __restrict__ wt2, const float* __restrict__ b2,
                                float* __restrict__ summed, int e_total) {
    extern __shared__ __align__(16) char smem[];
    int*      colIdx = (int*)smem;                          // 128
    int*      rowIdx = colIdx + TILE;                       // 128
    _Float16* Hs  = (_Float16*)(smem + 1024);
    _Float16* H1s = Hs  + TILE * PITCH;
    _Float16* W1s = H1s + TILE * PITCH;
    _Float16* W2s = W1s + DIM  * PITCH;

    const int tid   = threadIdx.x;
    const int ebase = blockIdx.x * TILE;

    // prefetch next tile's indices while we work on this one
    __builtin_prefetch(ei + ebase + TILE, 0, 0);
    __builtin_prefetch(ei + NEDGES + ebase + TILE, 0, 0);

    if (tid < TILE) {
        int e = ebase + tid;
        colIdx[tid] = (e < e_total) ? ei[NEDGES + e] : 0;
        rowIdx[tid] = (e < e_total) ? ei[e] : -1;
    }
    __syncthreads();

    // stage weights (with pitch padding) and gather activations, 16B chunks
    #pragma unroll 2
    for (int idx = tid; idx < DIM * 16; idx += 256) {
        int r = idx >> 4, c = idx & 15;
        async_cp16(W1s + r * PITCH + c * 8, wt1 + r * DIM + c * 8);
        async_cp16(W2s + r * PITCH + c * 8, wt2 + r * DIM + c * 8);
        async_cp16(Hs  + r * PITCH + c * 8,
                   xt + (size_t)colIdx[r] * DIM + c * 8);
    }
    async_join();
    __syncthreads();

    const int wave = tid >> 5, lane = tid & 31;
    const int l15 = lane & 15, hh = lane >> 4;
    const _Float16* Arow = Hs + (wave * 16 + l15) * PITCH;

    // ---- layer 1: relu(H @ W1 + b1) -> H1 (f16, wave-private strip) ----
    v8f acc[8];
    #pragma unroll
    for (int nt = 0; nt < 8; ++nt) acc[nt] = (v8f){0,0,0,0,0,0,0,0};
    #pragma unroll
    for (int kt = 0; kt < 4; ++kt) {
        v16h a = load_frag(Arow, kt, hh);
        #pragma unroll
        for (int nt = 0; nt < 8; ++nt) {
            v16h b = load_frag(W1s + (nt * 16 + l15) * PITCH, kt, hh);
            acc[nt] = __builtin_amdgcn_wmma_f32_16x16x32_f16(
                false, a, false, b, (short)0, acc[nt], false, false);
        }
    }
    #pragma unroll
    for (int nt = 0; nt < 8; ++nt) {
        float bn = b1[nt * 16 + l15];
        #pragma unroll
        for (int i = 0; i < 8; ++i) {
            float v = acc[nt][i] + bn;
            v = v > 0.0f ? v : 0.0f;
            H1s[(wave * 16 + i + 8 * hh) * PITCH + nt * 16 + l15] = (_Float16)v;
        }
    }
    // wave-private strip: per-wave LDS ordering suffices, no barrier needed

    // ---- layer 2: H1 @ W2 + b2, then atomic scatter-add ----
    const _Float16* A2row = H1s + (wave * 16 + l15) * PITCH;
    #pragma unroll
    for (int nt = 0; nt < 8; ++nt) acc[nt] = (v8f){0,0,0,0,0,0,0,0};
    #pragma unroll
    for (int kt = 0; kt < 4; ++kt) {
        v16h a = load_frag(A2row, kt, hh);
        #pragma unroll
        for (int nt = 0; nt < 8; ++nt) {
            v16h b = load_frag(W2s + (nt * 16 + l15) * PITCH, kt, hh);
            acc[nt] = __builtin_amdgcn_wmma_f32_16x16x32_f16(
                false, a, false, b, (short)0, acc[nt], false, false);
        }
    }
    #pragma unroll
    for (int i = 0; i < 8; ++i) {
        int m = wave * 16 + i + 8 * hh;           // local edge
        int r = rowIdx[m];                        // destination node
        if (r < 0) continue;
        float* dst = summed + (size_t)r * DIM + l15;
        #pragma unroll
        for (int nt = 0; nt < 8; ++nt) {
            float v = acc[nt][i] + b2[nt * 16 + l15];
            atomicAdd(dst + nt * 16, v);
        }
    }
}

// ---------------- node update: mean -> 2xGEMM -> expmap0 --------------------
__global__ void node_update_kernel(const float* __restrict__ summed,
                                   const float* __restrict__ cnt,
                                   const _Float16* __restrict__ wt1, const float* __restrict__ b1,
                                   const _Float16* __restrict__ wt2, const float* __restrict__ b2,
                                   float* __restrict__ out, int n) {
    extern __shared__ __align__(16) char smem[];
    float*    inv = (float*)smem;                           // 128
    _Float16* Xs  = (_Float16*)(smem + 512);
    _Float16* H1s = Xs  + TILE * PITCH;
    _Float16* W1s = H1s + TILE * PITCH;
    _Float16* W2s = W1s + DIM  * PITCH;
    float*    OutT = (float*)(smem + 512);                  // overlays Xs+H1s, pitch 132

    const int tid   = threadIdx.x;
    const int rbase = blockIdx.x * TILE;

    if (tid < TILE) {
        int g = rbase + tid;
        inv[tid] = (g < n) ? 1.0f / (cnt[g] + 1e-8f) : 0.0f;
    }
    #pragma unroll 2
    for (int idx = tid; idx < DIM * 16; idx += 256) {       // stage weights (async)
        int r = idx >> 4, c = idx & 15;
        async_cp16(W1s + r * PITCH + c * 8, wt1 + r * DIM + c * 8);
        async_cp16(W2s + r * PITCH + c * 8, wt2 + r * DIM + c * 8);
    }
    async_join();
    __syncthreads();
    // mean aggregate -> f16 LDS tile
    #pragma unroll 4
    for (int idx = tid; idx < TILE * 32; idx += 256) {
        int r = idx >> 5, c = idx & 31;
        int g = rbase + r;
        f4 v = (g < n) ? *(const f4*)(summed + (size_t)g * DIM + c * 4) : (f4){0,0,0,0};
        *(h4*)(Xs + r * PITCH + c * 4) = __builtin_convertvector(v * inv[r], h4);
    }
    __syncthreads();

    const int wave = tid >> 5, lane = tid & 31;
    const int l15 = lane & 15, hh = lane >> 4;

    v8f acc[8];
    #pragma unroll
    for (int nt = 0; nt < 8; ++nt) acc[nt] = (v8f){0,0,0,0,0,0,0,0};
    const _Float16* Arow = Xs + (wave * 16 + l15) * PITCH;
    #pragma unroll
    for (int kt = 0; kt < 4; ++kt) {
        v16h a = load_frag(Arow, kt, hh);
        #pragma unroll
        for (int nt = 0; nt < 8; ++nt) {
            v16h b = load_frag(W1s + (nt * 16 + l15) * PITCH, kt, hh);
            acc[nt] = __builtin_amdgcn_wmma_f32_16x16x32_f16(
                false, a, false, b, (short)0, acc[nt], false, false);
        }
    }
    #pragma unroll
    for (int nt = 0; nt < 8; ++nt) {
        float bn = b1[nt * 16 + l15];
        #pragma unroll
        for (int i = 0; i < 8; ++i) {
            float v = acc[nt][i] + bn;
            v = v > 0.0f ? v : 0.0f;
            H1s[(wave * 16 + i + 8 * hh) * PITCH + nt * 16 + l15] = (_Float16)v;
        }
    }
    const _Float16* A2row = H1s + (wave * 16 + l15) * PITCH;
    #pragma unroll
    for (int nt = 0; nt < 8; ++nt) acc[nt] = (v8f){0,0,0,0,0,0,0,0};
    #pragma unroll
    for (int kt = 0; kt < 4; ++kt) {
        v16h a = load_frag(A2row, kt, hh);
        #pragma unroll
        for (int nt = 0; nt < 8; ++nt) {
            v16h b = load_frag(W2s + (nt * 16 + l15) * PITCH, kt, hh);
            acc[nt] = __builtin_amdgcn_wmma_f32_16x16x32_f16(
                false, a, false, b, (short)0, acc[nt], false, false);
        }
    }
    __syncthreads();   // all waves done reading Xs/H1s before OutT overlay write

    #pragma unroll
    for (int nt = 0; nt < 8; ++nt) {
        float bn = b2[nt * 16 + l15];
        #pragma unroll
        for (int i = 0; i < 8; ++i)
            OutT[(wave * 16 + i + 8 * hh) * 132 + nt * 16 + l15] = acc[nt][i] + bn;
    }
    // expmap0 on wave-private rows (per-wave LDS ordering suffices)
    for (int rr = 0; rr < 16; ++rr) {
        int m = wave * 16 + rr;
        int g = rbase + m;
        f4 v = *(const f4*)(OutT + m * 132 + lane * 4);
        float ss = v.x * v.x + v.y * v.y + v.z * v.z + v.w * v.w;
        #pragma unroll
        for (int off = 16; off > 0; off >>= 1) ss += __shfl_xor(ss, off, 32);
        float nc = fmaxf(sqrtf(ss), 1e-8f);
        float sc = tanhf(nc) / nc;
        if (g < n) *(f4*)(out + (size_t)g * DIM + lane * 4) = v * sc;
    }
}

// ---------------- launch ----------------------------------------------------
extern "C" void kernel_launch(void* const* d_in, const int* in_sizes, int n_in,
                              void* d_out, int out_size, void* d_ws, size_t ws_size,
                              hipStream_t stream) {
    const float* x      = (const float*)d_in[0];
    const int*   ei     = (const int*)d_in[1];
    const float* w_msg1 = (const float*)d_in[2];
    const float* b_msg1 = (const float*)d_in[3];
    const float* w_msg2 = (const float*)d_in[4];
    const float* b_msg2 = (const float*)d_in[5];
    const float* w_upd1 = (const float*)d_in[6];
    const float* b_upd1 = (const float*)d_in[7];
    const float* w_upd2 = (const float*)d_in[8];
    const float* b_upd2 = (const float*)d_in[9];
    float* out = (float*)d_out;

    char* ws = (char*)d_ws;
    _Float16* xt = (_Float16*)ws;                       // N*128 f16
    size_t off = (size_t)NNODES * DIM * sizeof(_Float16);
    _Float16* wtm1 = (_Float16*)(ws + off);             // 4 x 128x128 f16 (transposed)
    _Float16* wtm2 = wtm1 + DIM * DIM;
    _Float16* wtu1 = wtm2 + DIM * DIM;
    _Float16* wtu2 = wtu1 + DIM * DIM;
    off += 4 * (size_t)DIM * DIM * sizeof(_Float16);
    float* summed = (float*)(ws + off);                 // N*128 f32
    off += (size_t)NNODES * DIM * sizeof(float);
    float* cnt = (float*)(ws + off);                    // N f32 (contiguous after summed)

    // zero summed + count in one shot (contiguous)
    (void)hipMemsetAsync(summed, 0, (size_t)NNODES * (DIM + 1) * sizeof(float), stream);

    wtrans_kernel<<<64, 256, 0, stream>>>(w_msg1, wtm1);
    wtrans_kernel<<<64, 256, 0, stream>>>(w_msg2, wtm2);
    wtrans_kernel<<<64, 256, 0, stream>>>(w_upd1, wtu1);
    wtrans_kernel<<<64, 256, 0, stream>>>(w_upd2, wtu2);

    logmap0_kernel<<<(NNODES + 7) / 8, 256, 0, stream>>>(x, xt, NNODES);
    degree_kernel<<<(NEDGES + 255) / 256, 256, 0, stream>>>(ei, cnt, NEDGES);

    size_t smemE = 1024 + 4u * TILE * PITCH * sizeof(_Float16);   // ~140 KB (WGP has 320 KB)
    edge_mlp_kernel<<<(NEDGES + TILE - 1) / TILE, 256, smemE, stream>>>(
        xt, ei, wtm1, b_msg1, wtm2, b_msg2, summed, NEDGES);

    size_t smemN = 512 + 4u * TILE * PITCH * sizeof(_Float16);
    node_update_kernel<<<(NNODES + TILE - 1) / TILE, 256, smemN, stream>>>(
        summed, cnt, wtu1, b_upd1, wtu2, b_upd2, out, NNODES);
}
